// _ElmoBiLm_36490042147341
// MI455X (gfx1250) — matrix operand order repaired and checked by reading, not compiled, not run on hardware
//
#include <hip/hip_runtime.h>

// ---------------------------------------------------------------------------
// ELMo BiLM forward for MI455X (gfx1250, wave32).
// - All matmuls are X @ W.T -> NT GEMM on v_wmma_f32_16x16x32_bf16 (f32 acc).
// - Weights are converted f32->bf16 ONCE (prologue); GEMM W-tiles are staged
//   into LDS via GLOBAL_LOAD_ASYNC_TO_LDS_B128 (ASYNCcnt) when available.
// - Activations (A tiles) are converted f32->bf16 (RNE) during LDS staging.
// ---------------------------------------------------------------------------

typedef __attribute__((ext_vector_type(16))) __bf16 v16bf;
typedef __attribute__((ext_vector_type(8)))  float  v8f;
typedef int v4i_vec __attribute__((vector_size(16)));   // int4 GCC-vector

#define AS1 __attribute__((address_space(1)))
#define AS3 __attribute__((address_space(3)))

__device__ __forceinline__ unsigned pack_bf2(float a, float b) {
    union { float f; unsigned u; } ua, ub;
    ua.f = a; ub.f = b;
    unsigned x = ua.u, y = ub.u;
    x = (x + 0x7FFFu + ((x >> 16) & 1u)) >> 16;   // round-to-nearest-even
    y = (y + 0x7FFFu + ((y >> 16) & 1u)) >> 16;
    return (x & 0xFFFFu) | (y << 16);
}

// 16 bytes global(bf16 weights) -> LDS, async (ASYNCcnt) if builtin available.
__device__ __forceinline__ void async_ld128(const unsigned short* g, unsigned* l) {
#if __has_builtin(__builtin_amdgcn_global_load_async_to_lds_b128)
    __builtin_amdgcn_global_load_async_to_lds_b128(
        (AS1 v4i_vec*)g, (AS3 v4i_vec*)l, 0, 0);
#else
    *(uint4*)l = *(const uint4*)g;
#endif
}

__device__ __forceinline__ void wait_async0() {
#if __has_builtin(__builtin_amdgcn_s_wait_asynccnt)
    __builtin_amdgcn_s_wait_asynccnt(0);
#elif defined(__gfx1250__)
    asm volatile("s_wait_asynccnt 0x0" ::: "memory");
#endif
}

// C[M,Ntot] (f32, stride ldc) = A[M,K] (f32, stride lda) @ Wbf[Ntot,K]^T (+bias)
// 256 threads = 8 waves in a WM x WN grid; each wave owns 32x32 (2x2 WMMA tiles).
template <int WM, int WN>
__global__ __launch_bounds__(256)
void gemm_bf16_nt(float* __restrict__ C, int ldc,
                  const float* __restrict__ A, int lda,
                  const unsigned short* __restrict__ W,  // bf16 [Ntot, K]
                  const float* __restrict__ bias,        // [Ntot] or nullptr
                  int M, int Ntot, int K)
{
    constexpr int BM = WM * 32;
    constexpr int BN = WN * 32;
    __shared__ unsigned sA[BM * 17];   // [row][kpair] stride 17 (bank spread)
    __shared__ unsigned sW[BN * 20];   // stride 20 -> rows stay 16B aligned

    const int tid  = threadIdx.x;
    const int lane = tid & 31;
    const int wv   = tid >> 5;
    const int wm   = wv / WN;
    const int wn   = wv % WN;
    const int hf   = lane >> 4;
    const int l16  = lane & 15;
    const int mBase = blockIdx.y * BM;
    const int nBase = blockIdx.x * BN;

    v8f acc[2][2];
#pragma unroll
    for (int s = 0; s < 2; ++s)
#pragma unroll
        for (int s2 = 0; s2 < 2; ++s2)
#pragma unroll
            for (int r = 0; r < 8; ++r) acc[s][s2][r] = 0.f;

    for (int k0 = 0; k0 < K; k0 += 32) {
        // ---- stage A tile (BM x 32), f32 -> bf16 on the fly ----
        for (int u = tid; u < BM * 16; u += 256) {
            int r = u >> 4, p = u & 15;
            int gm = mBase + r;
            unsigned v = 0u;
            if (gm < M) {
                const float* s = A + (size_t)gm * lda + k0 + p * 2;
                v = pack_bf2(s[0], s[1]);
                if (k0 + 32 < K) __builtin_prefetch(s + 32, 0, 1);  // global_prefetch_b8
            }
            sA[r * 17 + p] = v;
        }
        // ---- stage W tile (BN x 32 bf16): async global -> LDS, 16B chunks ----
        for (int u = tid; u < BN * 4; u += 256) {
            int r = u >> 2, c = u & 3;
            int gn = nBase + r;
            if (gn >= Ntot) gn = Ntot - 1;          // clamp (harmless duplicate)
            const unsigned short* src = W + (size_t)gn * K + k0 + c * 8;
            async_ld128(src, &sW[r * 20 + c * 4]);
        }
        wait_async0();          // my wave's async LDS writes have landed
        __syncthreads();        // everyone's tile is visible

        union Fr { unsigned u[8]; v16bf v; };
        Fr a[2], bf[2];
        // A frag: lane row m = l16; VGPR i holds k-pair p = (i&3)+4*hf+8*(i>>2)
#pragma unroll
        for (int s = 0; s < 2; ++s) {
            int mr = wm * 32 + s * 16 + l16;
#pragma unroll
            for (int i = 0; i < 8; ++i)
                a[s].u[i] = sA[mr * 17 + ((i & 3) + (hf << 2) + ((i >> 2) << 3))];
        }
        // B frag: lane col n = l16; VGPR j holds k-pair p = (j&3)+4*hf+8*(j>>2)
#pragma unroll
        for (int s2 = 0; s2 < 2; ++s2) {
            int nr = wn * 32 + s2 * 16 + l16;
#pragma unroll
            for (int j = 0; j < 8; ++j)
                bf[s2].u[j] = sW[nr * 20 + ((j & 3) + (hf << 2) + ((j >> 2) << 3))];
        }
#pragma unroll
        for (int s = 0; s < 2; ++s)
#pragma unroll
            for (int s2 = 0; s2 < 2; ++s2)
                acc[s][s2] = __builtin_amdgcn_wmma_f32_16x16x32_bf16(
                    false, a[s].v, false, bf[s2].v, (short)0, acc[s][s2],
                    false, false);
        __syncthreads();
    }

    // ---- store: D lane layout n = l16, m = 8*hf + r ----
#pragma unroll
    for (int s = 0; s < 2; ++s) {
#pragma unroll
        for (int s2 = 0; s2 < 2; ++s2) {
            int gn = nBase + wn * 32 + s2 * 16 + l16;
            float bv = (bias != nullptr && gn < Ntot) ? bias[gn] : 0.f;
#pragma unroll
            for (int r = 0; r < 8; ++r) {
                int gm = mBase + wm * 32 + s * 16 + hf * 8 + r;
                if (gm < M && gn < Ntot)
                    C[(size_t)gm * ldc + gn] = acc[s][s2][r] + bv;
            }
        }
    }
}

// one-time weight conversion: f32 -> packed bf16 pairs
__global__ void cvt_bf16(unsigned* __restrict__ dst, const float* __restrict__ src,
                         int npairs)
{
    int i = blockIdx.x * 256 + threadIdx.x;
    if (i < npairs) dst[i] = pack_bf2(src[2 * i], src[2 * i + 1]);
}

// ---------------------------------------------------------------------------
// Character-CNN token encoder: one token per block.
// ---------------------------------------------------------------------------
__global__ __launch_bounds__(256)
void tok_cnn(const int* __restrict__ ids, const float* __restrict__ emb,
             const float* w1, const float* b1, const float* w2, const float* b2,
             const float* w3, const float* b3, const float* w4, const float* b4,
             const float* w5, const float* b5, const float* w6, const float* b6,
             const float* w7, const float* b7, float* __restrict__ tok)
{
    const int T = 258;
    __shared__ float x[50 * 16];
    int n = blockIdx.x;
    int b = n / T, t = n % T;
    for (int u = threadIdx.x; u < 800; u += 256) {
        int c = u >> 4, j = u & 15;
        int id;
        if (t == 0 || t == T - 1) {
            id = (c == 0) ? 259 : (c == 1) ? (t == 0 ? 257 : 258)
               : (c == 2) ? 260 : 261;
        } else {
            id = ids[((size_t)b * (T - 2) + (t - 1)) * 50 + c];
        }
        x[u] = emb[id * 16 + j];
    }
    __syncthreads();

    for (int f = threadIdx.x; f < 2048; f += 256) {
        int wdt, lf; const float* W; const float* Bv;
        if      (f < 32)   { wdt = 1; lf = f;        W = w1; Bv = b1; }
        else if (f < 64)   { wdt = 2; lf = f - 32;   W = w2; Bv = b2; }
        else if (f < 128)  { wdt = 3; lf = f - 64;   W = w3; Bv = b3; }
        else if (f < 256)  { wdt = 4; lf = f - 128;  W = w4; Bv = b4; }
        else if (f < 512)  { wdt = 5; lf = f - 256;  W = w5; Bv = b5; }
        else if (f < 1024) { wdt = 6; lf = f - 512;  W = w6; Bv = b6; }
        else               { wdt = 7; lf = f - 1024; W = w7; Bv = b7; }
        const float* Wf = W + (size_t)lf * 16 * wdt;   // layout [n,16,w]
        float best = -3.4e38f;
        for (int p = 0; p <= 50 - wdt; ++p) {
            float s = 0.f;
            for (int j = 0; j < 16; ++j) {
                const float* wr = Wf + j * wdt;
                const float* xr = x + p * 16 + j;
                for (int q = 0; q < wdt; ++q) s += xr[q * 16] * wr[q];
            }
            best = fmaxf(best, s);
        }
        float o = best + Bv[lf];
        tok[(size_t)n * 2048 + f] = o > 0.f ? o : 0.f;
    }
}

// highway: h = g*h + (1-g)*relu(nl); p already contains h@W.T + bias from GEMM
__global__ void highway_ep(float* __restrict__ h, const float* __restrict__ p, int total)
{
    int idx = blockIdx.x * 256 + threadIdx.x;
    if (idx >= total) return;
    int n = idx >> 11, j = idx & 2047;
    const float* pr = p + (size_t)n * 4096;
    float nl = pr[j];
    float g  = pr[2048 + j];
    g  = 1.f / (1.f + __expf(-g));
    nl = nl > 0.f ? nl : 0.f;
    h[idx] = g * h[idx] + (1.f - g) * nl;
}

__global__ void write_acts0(float* __restrict__ out, const float* __restrict__ tokp, int total)
{
    int idx = blockIdx.x * 256 + threadIdx.x;
    if (idx >= total) return;
    int n = idx >> 9, j = idx & 511;
    float v = tokp[idx];
    out[(size_t)n * 1024 + j]       = v;
    out[(size_t)n * 1024 + 512 + j] = v;
}

__global__ void fillz(float* __restrict__ p, int n)
{
    int i = blockIdx.x * 256 + threadIdx.x;
    if (i < n) p[i] = 0.f;
}

__global__ __launch_bounds__(256)
void lstm_pw(const float* __restrict__ gates, const float* __restrict__ gin,
             const float* __restrict__ bias, float* __restrict__ c_state,
             float* __restrict__ m_in, int t, int T)
{
    int idx = blockIdx.x * 256 + threadIdx.x;     // < 32*4096
    int b = idx >> 12, u = idx & 4095;
    const float* gr = gates + (size_t)b * 16384;
    const float* gi = gin + ((size_t)b * T + t) * 16384;
    float vi = gr[u]         + gi[u]         + bias[u];
    float vf = gr[u + 4096]  + gi[u + 4096]  + bias[u + 4096];
    float vm = gr[u + 8192]  + gi[u + 8192]  + bias[u + 8192];
    float vo = gr[u + 12288] + gi[u + 12288] + bias[u + 12288];
    float si = 1.f / (1.f + __expf(-vi));
    float sf = 1.f / (1.f + __expf(-vf));
    float so = 1.f / (1.f + __expf(-vo));
    float c  = si * tanhf(vm) + sf * c_state[idx];
    c = fminf(3.f, fmaxf(-3.f, c));
    c_state[idx] = c;
    m_in[idx] = so * tanhf(c);
}

__global__ void lstm_fin(const float* __restrict__ hp, float* __restrict__ h_state,
                         float* __restrict__ out, int layer, int dir, int t,
                         int B, int T)
{
    int idx = blockIdx.x * 256 + threadIdx.x;     // < 32*512
    if (idx >= 32 * 512) return;
    int b = idx >> 9, u = idx & 511;
    float h = hp[idx];
    h = fminf(3.f, fmaxf(-3.f, h));
    h_state[idx] = h;
    out[(((size_t)(layer + 1) * B + b) * T + t) * 1024 + dir * 512 + u] = h;
}

__global__ void skip_add(float* __restrict__ dst, const float* __restrict__ src, int n)
{
    int i = blockIdx.x * 256 + threadIdx.x;
    if (i < n) dst[i] += src[i];
}

// ---------------------------------------------------------------------------
extern "C" void kernel_launch(void* const* d_in, const int* in_sizes, int n_in,
                              void* d_out_v, int out_size, void* d_ws, size_t ws_size,
                              hipStream_t stream)
{
    (void)in_sizes; (void)n_in; (void)out_size; (void)ws_size;
    const int B = 32, T = 258, N = B * T;   // 8256 tokens incl. boundaries

    const int*   ids    = (const int*)d_in[0];
    const float* emb    = (const float*)d_in[1];
    const float* cw[7], *cb[7];
    for (int i = 0; i < 7; ++i) { cw[i] = (const float*)d_in[2 + 2 * i];
                                  cb[i] = (const float*)d_in[3 + 2 * i]; }
    const float* hw_w1  = (const float*)d_in[16];
    const float* hw_b1  = (const float*)d_in[17];
    const float* hw_w2  = (const float*)d_in[18];
    const float* hw_b2  = (const float*)d_in[19];
    const float* proj_w = (const float*)d_in[20];
    const float* proj_b = (const float*)d_in[21];
    const float* Wi     = (const float*)d_in[22];   // [2,2,16384,512]
    const float* Wsr    = (const float*)d_in[23];   // [2,2,16384,512]
    const float* lb     = (const float*)d_in[24];   // [2,2,16384]
    const float* Wp     = (const float*)d_in[25];   // [2,2,512,4096]

    float* out = (float*)d_out_v;                   // [3, B, T, 1024]

    // ---- workspace: bf16 weights first (16B aligned regions), then f32 ----
    unsigned short* hw1b = (unsigned short*)d_ws;               //  8,388,608
    unsigned short* hw2b = hw1b + (size_t)8388608;              //  8,388,608
    unsigned short* prjb = hw2b + (size_t)8388608;              //  1,048,576
    unsigned short* Wib  = prjb + (size_t)1048576;              // 33,554,432
    unsigned short* Wsb  = Wib  + (size_t)33554432;             // 33,554,432
    unsigned short* Wpb  = Wsb  + (size_t)33554432;             //  8,388,608
    float* fs    = (float*)(Wpb + (size_t)8388608);
    float* tok   = fs;                              // [N, 2048]
    float* hwp   = tok   + (size_t)N * 2048;        // [N, 4096]
    float* tokp  = hwp   + (size_t)N * 4096;        // [N, 512]
    float* gin   = tokp  + (size_t)N * 512;         // [N, 16384] (reused per cell)
    float* gates = gin   + (size_t)N * 16384;       // [32, 16384]
    float* cst   = gates + (size_t)32 * 16384;      // [32, 4096]
    float* mint  = cst   + (size_t)32 * 4096;       // [32, 4096]
    float* hp    = mint  + (size_t)32 * 4096;       // [32, 512]
    float* hst   = hp    + (size_t)32 * 512;        // [32, 512]

    dim3 blk(256);
    const int mb = (N + 127) / 128;                 // 65 M-tiles for big GEMMs

    // 0) one-time weight conversion f32 -> bf16 (read once from HBM)
    cvt_bf16<<<(8388608 / 2 + 255) / 256, blk, 0, stream>>>((unsigned*)hw1b, hw_w1, 8388608 / 2);
    cvt_bf16<<<(8388608 / 2 + 255) / 256, blk, 0, stream>>>((unsigned*)hw2b, hw_w2, 8388608 / 2);
    cvt_bf16<<<(1048576 / 2 + 255) / 256, blk, 0, stream>>>((unsigned*)prjb, proj_w, 1048576 / 2);
    cvt_bf16<<<(33554432 / 2 + 255) / 256, blk, 0, stream>>>((unsigned*)Wib, Wi, 33554432 / 2);
    cvt_bf16<<<(33554432 / 2 + 255) / 256, blk, 0, stream>>>((unsigned*)Wsb, Wsr, 33554432 / 2);
    cvt_bf16<<<(8388608 / 2 + 255) / 256, blk, 0, stream>>>((unsigned*)Wpb, Wp, 8388608 / 2);

    // 1) char CNN -> tok [N, 2048]
    tok_cnn<<<N, blk, 0, stream>>>(ids, emb,
        cw[0], cb[0], cw[1], cb[1], cw[2], cb[2], cw[3], cb[3],
        cw[4], cb[4], cw[5], cb[5], cw[6], cb[6], tok);

    // 2) highway x2 (WMMA GEMM + gated epilogue)
    gemm_bf16_nt<4, 2><<<dim3(4096 / 64, mb), blk, 0, stream>>>(
        hwp, 4096, tok, 2048, hw1b, hw_b1, N, 4096, 2048);
    highway_ep<<<(N * 2048 + 255) / 256, blk, 0, stream>>>(tok, hwp, N * 2048);
    gemm_bf16_nt<4, 2><<<dim3(4096 / 64, mb), blk, 0, stream>>>(
        hwp, 4096, tok, 2048, hw2b, hw_b2, N, 4096, 2048);
    highway_ep<<<(N * 2048 + 255) / 256, blk, 0, stream>>>(tok, hwp, N * 2048);

    // 3) projection 2048 -> 512, then acts[0] = concat(tok, tok)
    gemm_bf16_nt<4, 2><<<dim3(512 / 64, mb), blk, 0, stream>>>(
        tokp, 512, tok, 2048, prjb, proj_b, N, 512, 2048);
    write_acts0<<<(N * 512 + 255) / 256, blk, 0, stream>>>(out, tokp, N * 512);

    // 4) 2-layer BiLSTM with projection + clipping
    for (int l = 0; l < 2; ++l) {
        for (int dir = 0; dir < 2; ++dir) {
            const float* Ax; int lda;
            if (l == 0) { Ax = tokp; lda = 512; }
            else        { Ax = out + (size_t)1 * B * T * 1024 + (size_t)dir * 512;
                          lda = 1024; }
            const unsigned short* Wic = Wib + (size_t)(l * 2 + dir) * 16384 * 512;
            const unsigned short* Wsc = Wsb + (size_t)(l * 2 + dir) * 16384 * 512;
            const float*          bc  = lb  + (size_t)(l * 2 + dir) * 16384;
            const unsigned short* Wpc = Wpb + (size_t)(l * 2 + dir) * 512 * 4096;

            // parallel input projection over all timesteps: gin = x @ Wi.T
            gemm_bf16_nt<4, 2><<<dim3(16384 / 64, mb), blk, 0, stream>>>(
                gin, 16384, Ax, lda, Wic, nullptr, N, 16384, 512);

            fillz<<<(32 * 512 + 255) / 256, blk, 0, stream>>>(hst, 32 * 512);
            fillz<<<(32 * 4096 + 255) / 256, blk, 0, stream>>>(cst, 32 * 4096);

            for (int st = 0; st < T; ++st) {
                int t = dir ? (T - 1 - st) : st;
                // recurrent: gates = h @ Ws.T  (M=32, Ws L2-resident)
                gemm_bf16_nt<1, 8><<<dim3(16384 / 256, 1), blk, 0, stream>>>(
                    gates, 16384, hst, 512, Wsc, nullptr, 32, 16384, 512);
                lstm_pw<<<(32 * 4096) / 256, blk, 0, stream>>>(
                    gates, gin, bc, cst, mint, t, T);
                // projection: hp = (o*tanh(c)) @ Wp.T
                gemm_bf16_nt<1, 8><<<dim3(512 / 256, 1), blk, 0, stream>>>(
                    hp, 512, mint, 4096, Wpc, nullptr, 32, 512, 4096);
                lstm_fin<<<(32 * 512 + 255) / 256, blk, 0, stream>>>(
                    hp, hst, out, l, dir, t, B, T);
            }
        }
    }

    // 5) skip connection on layer 2: acts[2] += acts[1]
    skip_add<<<(B * T * 1024 + 255) / 256, blk, 0, stream>>>(
        out + (size_t)2 * B * T * 1024, out + (size_t)1 * B * T * 1024,
        B * T * 1024);
}
